// MoE_LoRA_90013924590283
// MI455X (gfx1250) — matrix operand order repaired and checked by reading, not compiled, not run on hardware
//
#include <hip/hip_runtime.h>

#define T_TOK   4096
#define D_MODEL 768
#define D3      2304
#define H_HEADS 12
#define HD      64
#define S_SEQ   512
#define NEXP    8
#define LRANK   16
#define KLORA   128   // NEXP*LRANK

typedef __attribute__((ext_vector_type(16))) __bf16 v16bf;
typedef __attribute__((ext_vector_type(8)))  float  v8f;

union FragU { uint4 u[2]; v16bf v; };

// A-fragment (16x32 bf16, MxK): lane l -> row m = l&15.
// lanes 0-15 hold K {0..7, 16..23}; lanes 16-31 hold K {8..15, 24..31}.
__device__ __forceinline__ v16bf load_a_frag(const __bf16* X, int ld, int m0, int k0, int lane) {
  int row = m0 + (lane & 15);
  int kb  = k0 + ((lane >> 4) << 3);
  const __bf16* p = X + (size_t)row * ld + kb;
  FragU f;
  f.u[0] = *(const uint4*)(p);
  f.u[1] = *(const uint4*)(p + 16);
  return f.v;
}

// B-fragment (32x16 bf16, KxN) from row-major W[N,K]: lane l -> col n = l&15,
// lanes 0-15 hold K {0..7, 16..23}; lanes 16-31 hold K {8..15, 24..31}.
__device__ __forceinline__ v16bf load_b_frag(const __bf16* W, int ld, int n0, int k0, int lane) {
  int row = n0 + (lane & 15);
  int kb  = k0 + ((lane >> 4) << 3);
  const __bf16* p = W + (size_t)row * ld + kb;
  FragU f;
  f.u[0] = *(const uint4*)(p);
  f.u[1] = *(const uint4*)(p + 16);
  return f.v;
}

#define WMMA_BF16(a, b, c) \
  __builtin_amdgcn_wmma_f32_16x16x32_bf16(false, (a), false, (b), (short)0, (c), false, false)

// ---------------------------------------------------------------- converts
__global__ void cvt_f32_bf16(const float* __restrict__ in, __bf16* __restrict__ out, int n) {
  int i = blockIdx.x * blockDim.x + threadIdx.x;
  if (i < n) out[i] = (__bf16)in[i];
}

// B_lora [NEXP, 3D, RANK] -> B_cat [3D, NEXP*RANK], bf16
__global__ void build_bcat(const float* __restrict__ Bl, __bf16* __restrict__ bcat) {
  int i = blockIdx.x * blockDim.x + threadIdx.x;
  if (i < D3 * KLORA) {
    int o = i >> 7;            // output channel
    int c = i & (KLORA - 1);   // n*16 + r
    int n = c >> 4, r = c & 15;
    bcat[i] = (__bf16)Bl[((size_t)n * D3 + o) * LRANK + r];
  }
}

// ---------------------------------------------------------------- router
// one wave per token; disp[t][n] = topk ? prob * (ALPHA/RANK) : 0
__global__ __launch_bounds__(256) void router_kernel(
    const float* __restrict__ x, const float* __restrict__ Wg,
    const float* __restrict__ bg, float* __restrict__ disp) {
  int t = blockIdx.x * 8 + (threadIdx.x >> 5);
  int lane = threadIdx.x & 31;
  if (t >= T_TOK) return;
  const float* xr = x + (size_t)t * D_MODEL;
  float logit[NEXP];
  for (int n = 0; n < NEXP; ++n) {
    const float* w = Wg + n * D_MODEL;
    float s = 0.f;
    for (int k = lane; k < D_MODEL; k += 32) s += xr[k] * w[k];
    for (int m = 16; m >= 1; m >>= 1) s += __shfl_xor(s, m, 32);
    logit[n] = s + bg[n];
  }
  float mx = logit[0];
  for (int n = 1; n < NEXP; ++n) mx = fmaxf(mx, logit[n]);
  float p[NEXP], den = 0.f;
  for (int n = 0; n < NEXP; ++n) { p[n] = __expf(logit[n] - mx); den += p[n]; }
  float inv = 1.f / den;
  for (int n = 0; n < NEXP; ++n) p[n] *= inv;
  int i1 = 0;
  for (int n = 1; n < NEXP; ++n) if (p[n] > p[i1]) i1 = n;
  int i2 = (i1 == 0) ? 1 : 0;
  for (int n = 0; n < NEXP; ++n) if (n != i1 && p[n] > p[i2]) i2 = n;
  if (lane < NEXP) {
    float wv = (lane == i1 || lane == i2) ? p[lane] * 2.0f : 0.f;  // ALPHA/RANK = 2
    disp[t * NEXP + lane] = wv;
  }
}

// ---------------------------------------------------------------- GEMM core
// 32(M) x 64(N) wave tile, software-pipelined (double-buffered fragments).
__device__ __forceinline__ void gemm_tile_2x4(const __bf16* X, int ldx, const __bf16* W, int ldw,
                                              int m0, int n0, int K, int lane, v8f (&c)[8]) {
  v16bf a0 = load_a_frag(X, ldx, m0,      0, lane);
  v16bf a1 = load_a_frag(X, ldx, m0 + 16, 0, lane);
  v16bf b0 = load_b_frag(W, ldw, n0 +  0, 0, lane);
  v16bf b1 = load_b_frag(W, ldw, n0 + 16, 0, lane);
  v16bf b2 = load_b_frag(W, ldw, n0 + 32, 0, lane);
  v16bf b3 = load_b_frag(W, ldw, n0 + 48, 0, lane);
  for (int k0 = 32; k0 < K; k0 += 32) {
    // issue next k-step's loads before consuming current fragments
    v16bf na0 = load_a_frag(X, ldx, m0,      k0, lane);
    v16bf na1 = load_a_frag(X, ldx, m0 + 16, k0, lane);
    v16bf nb0 = load_b_frag(W, ldw, n0 +  0, k0, lane);
    v16bf nb1 = load_b_frag(W, ldw, n0 + 16, k0, lane);
    v16bf nb2 = load_b_frag(W, ldw, n0 + 32, k0, lane);
    v16bf nb3 = load_b_frag(W, ldw, n0 + 48, k0, lane);
    c[0] = WMMA_BF16(a0, b0, c[0]);
    c[1] = WMMA_BF16(a0, b1, c[1]);
    c[2] = WMMA_BF16(a0, b2, c[2]);
    c[3] = WMMA_BF16(a0, b3, c[3]);
    c[4] = WMMA_BF16(a1, b0, c[4]);
    c[5] = WMMA_BF16(a1, b1, c[5]);
    c[6] = WMMA_BF16(a1, b2, c[6]);
    c[7] = WMMA_BF16(a1, b3, c[7]);
    a0 = na0; a1 = na1; b0 = nb0; b1 = nb1; b2 = nb2; b3 = nb3;
  }
  c[0] = WMMA_BF16(a0, b0, c[0]);
  c[1] = WMMA_BF16(a0, b1, c[1]);
  c[2] = WMMA_BF16(a0, b2, c[2]);
  c[3] = WMMA_BF16(a0, b3, c[3]);
  c[4] = WMMA_BF16(a1, b0, c[4]);
  c[5] = WMMA_BF16(a1, b1, c[5]);
  c[6] = WMMA_BF16(a1, b2, c[6]);
  c[7] = WMMA_BF16(a1, b3, c[7]);
}

// ------------------------------------------------- LoRA down: a_w = (X A^T) .* disp
__global__ __launch_bounds__(256) void lora_down_kernel(
    const __bf16* __restrict__ X, const __bf16* __restrict__ Acat,
    const float* __restrict__ disp, __bf16* __restrict__ aw) {
  int wid = blockIdx.x * 8 + (threadIdx.x >> 5);
  int lane = threadIdx.x & 31;
  int tm = wid >> 1, tn = wid & 1;            // N = 128 -> 2 tiles of 64
  int m0 = tm * 32, n0 = tn * 64;
  v8f c[8] = {};
  gemm_tile_2x4(X, D_MODEL, Acat, D_MODEL, m0, n0, D_MODEL, lane, c);
  int lane15 = lane & 15, rhalf = (lane >> 4) << 3;
  for (int mi = 0; mi < 2; ++mi) {
    for (int j = 0; j < 4; ++j) {
      int n = n0 + j * 16 + lane15;
      int expert = n >> 4;
      for (int i = 0; i < 8; ++i) {
        int m = m0 + mi * 16 + i + rhalf;
        aw[(size_t)m * KLORA + n] = (__bf16)(c[mi * 4 + j][i] * disp[m * NEXP + expert]);
      }
    }
  }
}

// ------------------------------------------------- fused QKV + LoRA-up GEMM
// C = X Wqkv^T + a_w Bcat^T + bias ; scatter to q[BH,S,64], k[BH,S,64], v[BH,64,S]
__global__ __launch_bounds__(256) void qkv_kernel(
    const __bf16* __restrict__ X, const __bf16* __restrict__ Wq,
    const __bf16* __restrict__ aw, const __bf16* __restrict__ bcat,
    const float* __restrict__ bias,
    __bf16* __restrict__ qb, __bf16* __restrict__ kb, __bf16* __restrict__ vb) {
  int wid = blockIdx.x * 8 + (threadIdx.x >> 5);
  int lane = threadIdx.x & 31;
  int tm = wid / 36, tn = wid % 36;           // N = 2304 -> 36 tiles of 64
  int m0 = tm * 32, n0 = tn * 64;
  v8f c[8] = {};
  gemm_tile_2x4(X, D_MODEL, Wq, D_MODEL, m0, n0, D_MODEL, lane, c);
  gemm_tile_2x4(aw, KLORA, bcat, KLORA, m0, n0, KLORA, lane, c);
  int lane15 = lane & 15, rhalf = (lane >> 4) << 3;
  for (int mi = 0; mi < 2; ++mi) {
    for (int j = 0; j < 4; ++j) {
      int n = n0 + j * 16 + lane15;
      int which = n / D_MODEL;
      int rem = n % D_MODEL;
      int h = rem / HD, e = rem % HD;
      float bv = bias[n];
      for (int i = 0; i < 8; ++i) {
        int m = m0 + mi * 16 + i + rhalf;
        int b = m >> 9, s = m & (S_SEQ - 1);
        int bh = b * H_HEADS + h;
        float val = c[mi * 4 + j][i] + bv;
        if (which == 0)      qb[((size_t)bh * S_SEQ + s) * HD + e] = (__bf16)val;
        else if (which == 1) kb[((size_t)bh * S_SEQ + s) * HD + e] = (__bf16)val;
        else                 vb[((size_t)bh * HD + e) * S_SEQ + s] = (__bf16)val;  // V transposed
      }
    }
  }
}

// ------------------------------------------------- flash attention (per 16-query tile)
__global__ __launch_bounds__(256) void attn_kernel(
    const __bf16* __restrict__ qg, const __bf16* __restrict__ kg,
    const __bf16* __restrict__ vg, __bf16* __restrict__ x2) {
  __shared__ __bf16 plds[8 * 16 * 32];        // 1KB per wave P staging
  int warp = threadIdx.x >> 5;
  int lane = threadIdx.x & 31;
  int wid = blockIdx.x * 8 + warp;
  int bh = wid >> 5;                          // 32 query tiles per head
  int qtile = wid & 31;
  int b_idx = bh / H_HEADS, h_idx = bh % H_HEADS;
  int q0 = qtile * 16;

  const __bf16* qp = qg + (size_t)bh * S_SEQ * HD;
  const __bf16* kp = kg + (size_t)bh * S_SEQ * HD;
  const __bf16* vp = vg + (size_t)bh * HD * S_SEQ;

  v16bf aq0 = load_a_frag(qp, HD, q0, 0, lane);
  v16bf aq1 = load_a_frag(qp, HD, q0, 32, lane);

  v8f o0 = {}, o1 = {}, o2 = {}, o3 = {};
  float mx[8], sm[8];
  for (int i = 0; i < 8; ++i) { mx[i] = -1e30f; sm[i] = 0.f; }

  __bf16* pw = plds + warp * 512;
  int lane15 = lane & 15, rhalf = (lane >> 4) << 3;

  for (int jb = 0; jb < S_SEQ; jb += 32) {
    // scores: two 16x16 tiles (keys jb..jb+15, jb+16..jb+31), K-dim = 64
    v8f s0 = {}, s1 = {};
    s0 = WMMA_BF16(aq0, load_b_frag(kp, HD, jb, 0, lane), s0);
    s0 = WMMA_BF16(aq1, load_b_frag(kp, HD, jb, 32, lane), s0);
    s1 = WMMA_BF16(aq0, load_b_frag(kp, HD, jb + 16, 0, lane), s1);
    s1 = WMMA_BF16(aq1, load_b_frag(kp, HD, jb + 16, 32, lane), s1);

    // V fragments don't depend on scores: issue loads now, softmax hides latency
    v16bf bv0 = load_b_frag(vp, S_SEQ, 0,  jb, lane);
    v16bf bv1 = load_b_frag(vp, S_SEQ, 16, jb, lane);
    v16bf bv2 = load_b_frag(vp, S_SEQ, 32, jb, lane);
    v16bf bv3 = load_b_frag(vp, S_SEQ, 48, jb, lane);

    float cf[8];
    for (int i = 0; i < 8; ++i) {
      float a = s0[i] * 0.125f, cc = s1[i] * 0.125f;  // scale = hd^-0.5
      float r = fmaxf(a, cc);
      r = fmaxf(r, __shfl_xor(r, 1)); r = fmaxf(r, __shfl_xor(r, 2));
      r = fmaxf(r, __shfl_xor(r, 4)); r = fmaxf(r, __shfl_xor(r, 8));
      float mnew = fmaxf(mx[i], r);
      float corr = __expf(mx[i] - mnew);
      mx[i] = mnew;
      float p0 = __expf(a - mnew), p1 = __expf(cc - mnew);
      float rs = p0 + p1;
      rs += __shfl_xor(rs, 1); rs += __shfl_xor(rs, 2);
      rs += __shfl_xor(rs, 4); rs += __shfl_xor(rs, 8);
      sm[i] = sm[i] * corr + rs;
      cf[i] = corr;
      // stash P into LDS (C-layout row = i + 8*half, col = lane15 / 16+lane15)
      int row = i + rhalf;
      pw[row * 32 + lane15]      = (__bf16)p0;
      pw[row * 32 + 16 + lane15] = (__bf16)p1;
    }
    for (int i = 0; i < 8; ++i) { o0[i] *= cf[i]; o1[i] *= cf[i]; o2[i] *= cf[i]; o3[i] *= cf[i]; }

    // LDS ops from one wave stay in order; wait then reload P in A-layout
    asm volatile("s_wait_dscnt 0" ::: "memory");
    const __bf16* ap = pw + (size_t)(lane & 15) * 32 + ((lane >> 4) << 3);
    FragU f;
    f.u[0] = *(const uint4*)(ap);
    f.u[1] = *(const uint4*)(ap + 16);
    v16bf pa = f.v;

    o0 = WMMA_BF16(pa, bv0, o0);
    o1 = WMMA_BF16(pa, bv1, o1);
    o2 = WMMA_BF16(pa, bv2, o2);
    o3 = WMMA_BF16(pa, bv3, o3);
  }

  for (int i = 0; i < 8; ++i) {
    float inv = 1.f / sm[i];
    int srow = q0 + i + rhalf;
    size_t t = (size_t)b_idx * S_SEQ + srow;
    size_t col = (size_t)h_idx * HD + lane15;
    x2[t * D_MODEL + col +  0] = (__bf16)(o0[i] * inv);
    x2[t * D_MODEL + col + 16] = (__bf16)(o1[i] * inv);
    x2[t * D_MODEL + col + 32] = (__bf16)(o2[i] * inv);
    x2[t * D_MODEL + col + 48] = (__bf16)(o3[i] * inv);
  }
}

// ------------------------------------------------- output projection
__global__ __launch_bounds__(256) void proj_kernel(
    const __bf16* __restrict__ X, const __bf16* __restrict__ Wp,
    const float* __restrict__ bias, float* __restrict__ out) {
  int wid = blockIdx.x * 8 + (threadIdx.x >> 5);
  int lane = threadIdx.x & 31;
  int tm = wid / 12, tn = wid % 12;           // N = 768 -> 12 tiles of 64
  int m0 = tm * 32, n0 = tn * 64;
  v8f c[8] = {};
  gemm_tile_2x4(X, D_MODEL, Wp, D_MODEL, m0, n0, D_MODEL, lane, c);
  int lane15 = lane & 15, rhalf = (lane >> 4) << 3;
  for (int mi = 0; mi < 2; ++mi) {
    for (int j = 0; j < 4; ++j) {
      int n = n0 + j * 16 + lane15;
      float bv = bias[n];
      for (int i = 0; i < 8; ++i) {
        int m = m0 + mi * 16 + i + rhalf;
        out[(size_t)m * D_MODEL + n] = c[mi * 4 + j][i] + bv;
      }
    }
  }
}

// ---------------------------------------------------------------- launch
extern "C" void kernel_launch(void* const* d_in, const int* in_sizes, int n_in,
                              void* d_out, int out_size, void* d_ws, size_t ws_size,
                              hipStream_t stream) {
  const float* x    = (const float*)d_in[0];
  const float* Wqkv = (const float*)d_in[1];
  const float* bqkv = (const float*)d_in[2];
  const float* Wg   = (const float*)d_in[3];
  const float* bg   = (const float*)d_in[4];
  const float* A    = (const float*)d_in[5];   // [8,16,768] == Acat [128,768]
  const float* Bl   = (const float*)d_in[6];
  const float* Wp   = (const float*)d_in[7];
  const float* bp   = (const float*)d_in[8];
  float* out = (float*)d_out;

  char* ws = (char*)d_ws;
  size_t off = 0;
  auto carve = [&](size_t bytes) -> void* {
    void* p = ws + off;
    off += (bytes + 255) & ~(size_t)255;
    return p;
  };
  __bf16* xb     = (__bf16*)carve((size_t)T_TOK * D_MODEL * 2);
  __bf16* wqkvb  = (__bf16*)carve((size_t)D3 * D_MODEL * 2);
  __bf16* acatb  = (__bf16*)carve((size_t)KLORA * D_MODEL * 2);
  __bf16* bcatb  = (__bf16*)carve((size_t)D3 * KLORA * 2);
  __bf16* wprojb = (__bf16*)carve((size_t)D_MODEL * D_MODEL * 2);
  float*  disp   = (float*)carve((size_t)T_TOK * NEXP * 4);
  __bf16* aw     = (__bf16*)carve((size_t)T_TOK * KLORA * 2);
  __bf16* qb     = (__bf16*)carve((size_t)H_HEADS * 8 * S_SEQ * HD * 2);
  __bf16* kb     = (__bf16*)carve((size_t)H_HEADS * 8 * S_SEQ * HD * 2);
  __bf16* vb     = (__bf16*)carve((size_t)H_HEADS * 8 * S_SEQ * HD * 2);
  __bf16* x2     = (__bf16*)carve((size_t)T_TOK * D_MODEL * 2);

  const int thr = 256;
  int n;
  n = T_TOK * D_MODEL;
  cvt_f32_bf16<<<(n + thr - 1) / thr, thr, 0, stream>>>(x, xb, n);
  n = D3 * D_MODEL;
  cvt_f32_bf16<<<(n + thr - 1) / thr, thr, 0, stream>>>(Wqkv, wqkvb, n);
  n = KLORA * D_MODEL;
  cvt_f32_bf16<<<(n + thr - 1) / thr, thr, 0, stream>>>(A, acatb, n);
  n = D_MODEL * D_MODEL;
  cvt_f32_bf16<<<(n + thr - 1) / thr, thr, 0, stream>>>(Wp, wprojb, n);
  n = D3 * KLORA;
  build_bcat<<<(n + thr - 1) / thr, thr, 0, stream>>>(Bl, bcatb);

  router_kernel<<<T_TOK / 8, thr, 0, stream>>>(x, Wg, bg, disp);

  // (T/32) * (128/64) = 256 wave-tiles -> 32 blocks of 8 waves
  lora_down_kernel<<<32, thr, 0, stream>>>(xb, acatb, disp, aw);

  // (T/32) * (2304/64) = 4608 wave-tiles -> 576 blocks
  qkv_kernel<<<576, thr, 0, stream>>>(xb, wqkvb, aw, bcatb, bqkv, qb, kb, vb);

  // 96 heads * 32 query tiles = 3072 wave-tiles -> 384 blocks
  attn_kernel<<<384, thr, 0, stream>>>(qb, kb, vb, x2);

  // (T/32) * (768/64) = 1536 wave-tiles -> 192 blocks
  proj_kernel<<<192, thr, 0, stream>>>(x2, wprojb, bp, out);
}
